// GraphUnet_26302379721323
// MI455X (gfx1250) — compile-verified, matrix operand order checked by reading, not run on hardware
//
#include <hip/hip_runtime.h>
#include <stdint.h>

// GraphUnet on MI455X (gfx1250, wave32).
// All (A@X)@W+b GCN layers run through V_WMMA_F32_16X16X32_BF16 with fp32
// accumulation; A is converted to bf16 once per pyramid level (memory-bound
// regime: AI = 64 flop/byte vs 23.3 TB/s HBM). Tile feeds use
// GLOBAL_LOAD_ASYNC_TO_LDS_B128 (ASYNCcnt-tracked, no VGPR staging).
// Workspace use: ~340 MB.

typedef __bf16 bf16;
typedef __bf16 v16bf __attribute__((ext_vector_type(16)));
typedef float  v8f   __attribute__((ext_vector_type(8)));

static __device__ __forceinline__ bf16 f2bf(float f) {
  union { float f; uint32_t u; } a; a.f = f;
  uint32_t r = (a.u + 0x7FFFu + ((a.u >> 16) & 1u)) >> 16;
  union { uint16_t s; bf16 b; } o; o.s = (uint16_t)r;
  return o.b;
}

// ---- CDNA5 async global->LDS (b128) ----------------------------------------
#if __has_builtin(__builtin_amdgcn_global_load_async_to_lds_b128)
#define HAVE_ASYNC_LDS 1
typedef int v4i __attribute__((vector_size(16)));
typedef __attribute__((address_space(1))) v4i gas_v4i;  // global (AS1)
typedef __attribute__((address_space(3))) v4i las_v4i;  // LDS (AS3)
static __device__ __forceinline__ void async_b128(const void* g, void* l) {
  __builtin_amdgcn_global_load_async_to_lds_b128((gas_v4i*)g, (las_v4i*)l,
                                                 0, 0);
}
template <int N>
static __device__ __forceinline__ void wait_async() {
#if __has_builtin(__builtin_amdgcn_s_wait_asynccnt)
  __builtin_amdgcn_s_wait_asynccnt(N);
#else
  asm volatile("s_wait_asynccnt %0" ::"n"(N) : "memory");
#endif
}
#else
#define HAVE_ASYNC_LDS 0
#endif

// ---------------- A (fp32 n x n) -> bf16 (np x np, zero padded) -------------
__global__ void k_convA(const float* __restrict__ A, int n, int np_,
                        bf16* __restrict__ Abf) {
  int c0 = (blockIdx.x * blockDim.x + threadIdx.x) * 8;
  int r  = blockIdx.y;
  if (c0 >= np_) return;
  __align__(16) bf16 v[8];
#pragma unroll
  for (int j = 0; j < 8; ++j) {
    int c = c0 + j;
    float f = (r < n && c < n) ? A[(size_t)r * n + c] : 0.0f;
    v[j] = f2bf(f);
  }
  *(uint4*)&Abf[(size_t)r * np_ + c0] = *(uint4*)v;
}

// ---------------- X (fp32 n x 64) -> bf16 (np x 64, zero padded) ------------
__global__ void k_convX(const float* __restrict__ X, int n, int np_,
                        bf16* __restrict__ Xbf) {
  int t = blockIdx.x * 256 + threadIdx.x;
  int r = t >> 6, d = t & 63;
  if (r >= np_) return;
  float f = (r < n) ? X[(size_t)r * 64 + d] : 0.0f;
  Xbf[(size_t)r * 64 + d] = f2bf(f);
}

// ---------------- transpose Xbf (np x DIMX) -> XT (DIMX x np) ---------------
template <int DIMX>
__global__ void k_xt(const bf16* __restrict__ Xbf, int np_,
                     bf16* __restrict__ XT) {
  int r = blockIdx.x * 256 + threadIdx.x;
  int d = blockIdx.y;
  if (r >= np_) return;
  XT[(size_t)d * np_ + r] = Xbf[(size_t)r * DIMX + d];
}

// ---------------- fused GCN: Y = (A @ X) @ W + b (+resid) -------------------
// A: bf16 np x np (lda), XT: bf16 DIMX x kp (K-major), W: fp32 DIMX x 64.
// 256 threads = 8 wave32 waves; block computes 64 output rows, all 64 cols.
template <int DIMX>
__global__ __launch_bounds__(256) void k_gcn(
    const bf16* __restrict__ A, int lda,
    const bf16* __restrict__ XT, int kp,
    const float* __restrict__ W,
    const float* __restrict__ bias,
    const float* __restrict__ resid,
    float* __restrict__ Yf,
    bf16* __restrict__ Ybf) {
  constexpr int NT  = DIMX / 16;   // N tiles
  constexpr int TPW = NT / 2;      // C tiles per wave
  constexpr int NB  = DIMX / 32;   // XT row groups per thread in loader
  constexpr int LDA = 72;          // LDS K-stride (64 + pad)
  constexpr int PLD = DIMX + 4;
  constexpr int AS_ELEMS = 64 * LDA;
  constexpr int BS_ELEMS = DIMX * LDA;
  constexpr int AB_BYTES = 2 * (AS_ELEMS + BS_ELEMS) * (int)sizeof(bf16);
  constexpr int PL_BYTES = 64 * PLD * (int)sizeof(float);
  constexpr int R0 = AB_BYTES > PL_BYTES ? AB_BYTES : PL_BYTES;
  constexpr int WL_BYTES = (DIMX == 64) ? 64 * 64 * (int)sizeof(float) : 0;

  __shared__ __align__(16) char smem[R0 + WL_BYTES];
  bf16*  As = (bf16*)smem;             // [2][AS_ELEMS]
  bf16*  Bs = As + 2 * AS_ELEMS;       // [2][BS_ELEMS]
  float* Pl = (float*)smem;            // overlaps As/Bs (used after main loop)
  float* Wl = (float*)(smem + R0);     // staged W (DIMX==64 only)

  const int tid = threadIdx.x;
  const int blockRow = blockIdx.x * 64;

  if constexpr (DIMX == 64) {
    for (int i = tid; i < DIMX * 64; i += 256) Wl[i] = W[i];
  }

  const int lr = tid >> 3;            // 0..31
  const int g8 = (tid & 7) * 8;       // 16B column group
  const bf16* Ag0 = A + (size_t)(blockRow + lr) * lda + g8;
  const bf16* Ag1 = Ag0 + (size_t)32 * lda;

  const int lane = tid & 31;
  const int w    = tid >> 5;
  const int mi   = w & 3;
  const int nb   = (w >> 2) * TPW;
  const int mrow = mi * 16 + (lane & 15);
  const int kh   = (lane >> 4) * 8;

  const v8f vzero = {0.f, 0.f, 0.f, 0.f, 0.f, 0.f, 0.f, 0.f};
  v8f acc[TPW];
#pragma unroll
  for (int t = 0; t < TPW; ++t) acc[t] = vzero;

  const int nc = kp / 64;

#if HAVE_ASYNC_LDS
  // ---- async pipeline: global -> LDS directly, ASYNCcnt-tracked ----
  auto issue = [&](int kc, int buf) {
    const size_t ko = (size_t)kc * 64;
    bf16* a = As + buf * AS_ELEMS;
    bf16* bsh = Bs + buf * BS_ELEMS;
    async_b128(Ag0 + ko, &a[lr * LDA + g8]);
    async_b128(Ag1 + ko, &a[(lr + 32) * LDA + g8]);
#pragma unroll
    for (int ii = 0; ii < NB; ++ii)
      async_b128(XT + (size_t)(lr + 32 * ii) * kp + ko + g8,
                 &bsh[(lr + 32 * ii) * LDA + g8]);
  };
  issue(0, 0);
  for (int c = 0; c < nc; ++c) {
    const int buf = c & 1;
    if (c + 1 < nc) {
      issue(c + 1, buf ^ 1);
      wait_async<2 + NB>();  // chunk c's (2+NB) ops retired; c+1 in flight
    } else {
      wait_async<0>();
    }
    __syncthreads();
    const bf16* a = As + buf * AS_ELEMS;
    const bf16* b = Bs + buf * BS_ELEMS;
#pragma unroll
    for (int ks = 0; ks < 2; ++ks) {
      const int kb = ks * 32 + kh;
      union { v16bf v; uint4 q[2]; } af;
      af.q[0] = *(const uint4*)&a[mrow * LDA + kb];
      af.q[1] = *(const uint4*)&a[mrow * LDA + kb + 16];
#pragma unroll
      for (int t = 0; t < TPW; ++t) {
        const int ncol = (nb + t) * 16 + (lane & 15);
        union { v16bf v; uint4 q[2]; } bfr;
        bfr.q[0] = *(const uint4*)&b[ncol * LDA + kb];
        bfr.q[1] = *(const uint4*)&b[ncol * LDA + kb + 16];
        acc[t] = __builtin_amdgcn_wmma_f32_16x16x32_bf16(
            false, af.v, false, bfr.v, (short)0, acc[t], false, false);
      }
    }
    __syncthreads();  // everyone done reading buf before c+2 overwrites it
  }
#else
  // ---- fallback: synchronous loads, named scalars (no spillable arrays) ----
  for (int c = 0; c < nc; ++c) {
    const size_t ko = (size_t)c * 64;
    uint4 ra0 = *(const uint4*)(Ag0 + ko);
    uint4 ra1 = *(const uint4*)(Ag1 + ko);
    uint4 rb0 = *(const uint4*)(XT + (size_t)lr * kp + ko + g8);
    uint4 rb1 = *(const uint4*)(XT + (size_t)(lr + 32) * kp + ko + g8);
    uint4 rb2, rb3;
    if constexpr (NB == 4) {
      rb2 = *(const uint4*)(XT + (size_t)(lr + 64) * kp + ko + g8);
      rb3 = *(const uint4*)(XT + (size_t)(lr + 96) * kp + ko + g8);
    }
    __syncthreads();
    *(uint4*)&As[lr * LDA + g8] = ra0;
    *(uint4*)&As[(lr + 32) * LDA + g8] = ra1;
    *(uint4*)&Bs[lr * LDA + g8] = rb0;
    *(uint4*)&Bs[(lr + 32) * LDA + g8] = rb1;
    if constexpr (NB == 4) {
      *(uint4*)&Bs[(lr + 64) * LDA + g8] = rb2;
      *(uint4*)&Bs[(lr + 96) * LDA + g8] = rb3;
    }
    __syncthreads();
#pragma unroll
    for (int ks = 0; ks < 2; ++ks) {
      const int kb = ks * 32 + kh;
      union { v16bf v; uint4 q[2]; } af;
      af.q[0] = *(const uint4*)&As[mrow * LDA + kb];
      af.q[1] = *(const uint4*)&As[mrow * LDA + kb + 16];
#pragma unroll
      for (int t = 0; t < TPW; ++t) {
        const int ncol = (nb + t) * 16 + (lane & 15);
        union { v16bf v; uint4 q[2]; } bfr;
        bfr.q[0] = *(const uint4*)&Bs[ncol * LDA + kb];
        bfr.q[1] = *(const uint4*)&Bs[ncol * LDA + kb + 16];
        acc[t] = __builtin_amdgcn_wmma_f32_16x16x32_bf16(
            false, af.v, false, bfr.v, (short)0, acc[t], false, false);
      }
    }
  }
#endif

  __syncthreads();  // done reading As/Bs; Pl overlaps them
  {
    const int prow = mi * 16 + (lane >> 4) * 8;
    const int pc   = lane & 15;
#pragma unroll
    for (int t = 0; t < TPW; ++t) {
      const int pcol = (nb + t) * 16 + pc;
#pragma unroll
      for (int j = 0; j < 8; ++j) Pl[(prow + j) * PLD + pcol] = acc[t][j];
    }
  }
  __syncthreads();

  // fp32 epilogue: Y[r][c] = sum_d P[r][d] * W[d][c] + b[c] (+resid)
  const int r  = tid >> 2;
  const int c0 = (tid & 3) * 16;
  const float* Wep = (DIMX == 64) ? Wl : W;
  float o[16];
#pragma unroll
  for (int j = 0; j < 16; ++j) o[j] = bias[c0 + j];
  for (int d = 0; d < DIMX; ++d) {
    const float p = Pl[r * PLD + d];
#pragma unroll
    for (int j = 0; j < 16; ++j) o[j] += p * Wep[d * 64 + c0 + j];
  }
  const size_t row = (size_t)(blockRow + r);
  if (resid) {
#pragma unroll
    for (int j = 0; j < 16; ++j) o[j] += resid[row * 64 + c0 + j];
  }
  if (Yf) {
#pragma unroll
    for (int j = 0; j < 16; ++j) Yf[row * 64 + c0 + j] = o[j];
  }
  if (Ybf) {
    __align__(16) bf16 ob[16];
#pragma unroll
    for (int j = 0; j < 16; ++j) ob[j] = f2bf(o[j]);
    *(uint4*)&Ybf[row * 64 + c0]     = *(uint4*)&ob[0];
    *(uint4*)&Ybf[row * 64 + c0 + 8] = *(uint4*)&ob[8];
  }
}

// ---------------- pool scores: sigmoid((X @ Wp + bp) / 100) -----------------
__global__ void k_scores(const float* __restrict__ X, int n,
                         const float* __restrict__ Wp,
                         const float* __restrict__ bp,
                         float* __restrict__ s) {
  int r = blockIdx.x * 256 + threadIdx.x;
  if (r >= n) return;
  float z = bp[0];
#pragma unroll
  for (int d = 0; d < 64; ++d) z += X[(size_t)r * 64 + d] * Wp[d];
  z *= 0.01f;
  s[r] = 1.0f / (1.0f + expf(-z));
}

// ---------------- exact stable top-k via rank counting ----------------------
// rank(i) = #{ j : s[j] > s[i]  or  (s[j] == s[i] and j < i) }  (== lax.top_k)
__global__ void k_topk(const float* __restrict__ s, int n, int k,
                       int* __restrict__ idx, float* __restrict__ val) {
  const int i = blockIdx.x;
  const float si = s[i];
  __shared__ int red[256];
  int cnt = 0;
  for (int j = threadIdx.x; j < n; j += 256) {
    float sj = s[j];
    cnt += (sj > si) || (sj == si && j < i);
  }
  red[threadIdx.x] = cnt;
  __syncthreads();
  for (int off = 128; off; off >>= 1) {
    if (threadIdx.x < off) red[threadIdx.x] += red[threadIdx.x + off];
    __syncthreads();
  }
  if (threadIdx.x == 0) {
    int r = red[0];
    if (r < k) { idx[r] = i; val[r] = si; }
  }
}

// ---------------- pooled X: Xnew[r] = Xold[idx[r]] * val[r] (bf16, padded) --
__global__ void k_poolX(const float* __restrict__ Xold,
                        const int* __restrict__ idx,
                        const float* __restrict__ val, int k,
                        bf16* __restrict__ Xb) {
  int r = blockIdx.x, d = threadIdx.x;
  float v = 0.0f;
  if (r < k) v = Xold[(size_t)idx[r] * 64 + d] * val[r];
  Xb[(size_t)r * 64 + d] = f2bf(v);
}

// ---------------- pooled A: Anew[r][c] = Aold[idx[r]][idx[c]] (bf16) --------
__global__ void k_poolA(const bf16* __restrict__ Aold, int ldo,
                        const int* __restrict__ idx, int k, int kp,
                        bf16* __restrict__ Anew) {
  int c = blockIdx.x * 256 + threadIdx.x;
  int r = blockIdx.y;
  if (c >= kp) return;
  bf16 v = f2bf(0.0f);
  if (r < k && c < k) v = Aold[(size_t)idx[r] * ldo + idx[c]];
  Anew[(size_t)r * kp + c] = v;
}

__global__ void k_zero(uint32_t* __restrict__ p, long n) {
  long t = (long)blockIdx.x * 256 + threadIdx.x;
  if (t < n) p[t] = 0u;
}

// ---------------- unpool scatter: Xn[idx[r]] = Xsm[r] (bf16, pre-zeroed) ----
__global__ void k_unpool(const float* __restrict__ Xsm,
                         const int* __restrict__ idx, int k,
                         bf16* __restrict__ Xb) {
  int r = blockIdx.x, d = threadIdx.x;
  if (r >= k) return;
  Xb[(size_t)idx[r] * 64 + d] = f2bf(Xsm[(size_t)r * 64 + d]);
}

// ---------------- concat [X, orgX] directly into K-major XT (128 x np) ------
__global__ void k_concatT(const bf16* __restrict__ Xa,
                          const bf16* __restrict__ Xo, int np_,
                          bf16* __restrict__ XT) {
  int r = blockIdx.x * 256 + threadIdx.x;
  int c = blockIdx.y;  // 0..127
  if (r >= np_) return;
  bf16 v = (c < 64) ? Xa[(size_t)r * 64 + c] : Xo[(size_t)r * 64 + (c - 64)];
  XT[(size_t)c * np_ + r] = v;
}

// ============================================================================
extern "C" void kernel_launch(void* const* d_in, const int* in_sizes, int n_in,
                              void* d_out, int out_size, void* d_ws,
                              size_t ws_size, hipStream_t stream) {
  (void)in_sizes; (void)n_in; (void)out_size; (void)ws_size;
  const float* A       = (const float*)d_in[0];
  const float* Xin     = (const float*)d_in[1];
  const float* W_start = (const float*)d_in[2];
  const float* b_start = (const float*)d_in[3];
  const float* W_down  = (const float*)d_in[4];
  const float* b_down  = (const float*)d_in[5];
  const float* W_pool  = (const float*)d_in[6];
  const float* b_pool  = (const float*)d_in[7];
  const float* W_bot   = (const float*)d_in[8];
  const float* b_bot   = (const float*)d_in[9];
  const float* W_up    = (const float*)d_in[10];
  const float* b_up    = (const float*)d_in[11];
  const float* W_end   = (const float*)d_in[12];
  const float* b_end   = (const float*)d_in[13];

  // level sizes (k = int(KS[i]*n), matches Python float semantics) + pad to 64
  static const int nn[5] = {8192, 7372, 5160, 3096, 1548};
  static const int np[5] = {8192, 7424, 5184, 3136, 1600};

  char* ws = (char*)d_ws;
  auto alloc = [&](size_t bytes) -> void* {
    void* p = (void*)ws; ws += (bytes + 255) & ~(size_t)255; return p;
  };

  bf16* Abf[5];
  for (int l = 0; l < 5; ++l)
    Abf[l] = (bf16*)alloc((size_t)np[l] * np[l] * sizeof(bf16));
  int* idxs[4]; float* vals[4];
  for (int i = 0; i < 4; ++i) {
    idxs[i] = (int*)alloc((size_t)np[i + 1] * 4);
    vals[i] = (float*)alloc((size_t)np[i + 1] * 4);
  }
  float* scores = (float*)alloc((size_t)np[0] * 4);
  float* Xf_b   = (float*)alloc((size_t)np[0] * 64 * 4);  // fp32 up-chain
  bf16*  Xbf    = (bf16*)alloc((size_t)np[0] * 64 * 2);
  bf16*  XT     = (bf16*)alloc((size_t)64 * np[0] * 2);
  float* down_f[4];
  for (int i = 0; i < 4; ++i)
    down_f[i] = (float*)alloc((size_t)np[i] * 64 * 4);
  bf16* orgXbf = (bf16*)alloc((size_t)np[0] * 64 * 2);
  bf16* orgXT  = (bf16*)alloc((size_t)64 * np[0] * 2);
  bf16* XT128  = (bf16*)alloc((size_t)128 * np[0] * 2);

  float* out_final = (float*)d_out;                   // [8192,64]
  float* out_start = out_final + (size_t)8192 * 64;   // [8192,64]

  // A0 -> bf16
  {
    dim3 gr((np[0] / 8 + 255) / 256, np[0]);
    k_convA<<<gr, 256, 0, stream>>>(A, nn[0], np[0], Abf[0]);
  }
  // X -> bf16 -> XT
  k_convX<<<(np[0] * 64 + 255) / 256, 256, 0, stream>>>(Xin, nn[0], np[0], Xbf);
  { dim3 gr((np[0] + 255) / 256, 64);
    k_xt<64><<<gr, 256, 0, stream>>>(Xbf, np[0], XT); }

  // start GCN -> d_out (second tuple element) + orgXbf
  k_gcn<64><<<np[0] / 64, 256, 0, stream>>>(Abf[0], np[0], XT, np[0],
      W_start, b_start, nullptr, out_start, orgXbf);
  { dim3 gr((np[0] + 255) / 256, 64);
    k_xt<64><<<gr, 256, 0, stream>>>(orgXbf, np[0], orgXT); }

  // down path
  for (int i = 0; i < 4; ++i) {
    const int nl = nn[i], npl = np[i], k = nn[i + 1], kp = np[i + 1];
    const bf16* XTin = (i == 0) ? orgXT : XT;
    k_gcn<64><<<npl / 64, 256, 0, stream>>>(Abf[i], npl, XTin, npl,
        W_down + (size_t)i * 64 * 64, b_down + i * 64, nullptr,
        down_f[i], nullptr);
    k_scores<<<(nl + 255) / 256, 256, 0, stream>>>(down_f[i], nl,
        W_pool + i * 64, b_pool + i, scores);
    k_topk<<<nl, 256, 0, stream>>>(scores, nl, k, idxs[i], vals[i]);
    k_poolX<<<kp, 64, 0, stream>>>(down_f[i], idxs[i], vals[i], k, Xbf);
    { dim3 gr((kp + 255) / 256, 64);
      k_xt<64><<<gr, 256, 0, stream>>>(Xbf, kp, XT); }
    { dim3 gr((kp + 255) / 256, kp);
      k_poolA<<<gr, 256, 0, stream>>>(Abf[i], npl, idxs[i], k, kp, Abf[i + 1]); }
  }

  // bottom GCN
  k_gcn<64><<<np[4] / 64, 256, 0, stream>>>(Abf[4], np[4], XT, np[4],
      W_bot, b_bot, nullptr, Xf_b, nullptr);

  // up path (i indexes W_up/b_up, up indexes A/idx/residual; matches ref)
  for (int i = 0; i < 4; ++i) {
    const int up = 3 - i;
    const int npl = np[up], k = nn[up + 1];
    const long zb = (long)npl * 64 / 2;  // bf16 elements as u32 pairs
    k_zero<<<(unsigned)((zb + 255) / 256), 256, 0, stream>>>((uint32_t*)Xbf, zb);
    k_unpool<<<k, 64, 0, stream>>>(Xf_b, idxs[up], k, Xbf);
    { dim3 gr((npl + 255) / 256, 64);
      k_xt<64><<<gr, 256, 0, stream>>>(Xbf, npl, XT); }
    const bool last = (i == 3);
    k_gcn<64><<<npl / 64, 256, 0, stream>>>(Abf[up], npl, XT, npl,
        W_up + (size_t)i * 64 * 64, b_up + i * 64, down_f[up],
        Xf_b, last ? Xbf : nullptr);
  }

  // concat [X, orgX] -> XT128, final GCN -> d_out (first tuple element)
  { dim3 gr((np[0] + 255) / 256, 128);
    k_concatT<<<gr, 256, 0, stream>>>(Xbf, orgXbf, np[0], XT128); }
  k_gcn<128><<<np[0] / 64, 256, 0, stream>>>(Abf[0], np[0], XT128, np[0],
      W_end, b_end, nullptr, out_final, nullptr);
}
